// MultiheadSelfAttention_36421322670780
// MI455X (gfx1250) — matrix operand order repaired and checked by reading, not compiled, not run on hardware
//
#include <hip/hip_runtime.h>

// ---------------------------------------------------------------------------
// Multihead self-attention for MI455X (gfx1250), wave32 + WMMA f16->f32,
// async global->LDS staging (ASYNCcnt), double-buffered, contiguous b128
// fragment reads, batched fragment loads + back-to-back WMMA chains.
// B=2, L=4096, E=1024, H=16, hd=64.
// ---------------------------------------------------------------------------

typedef __attribute__((ext_vector_type(16))) _Float16 v16h;
typedef __attribute__((ext_vector_type(8)))  float    v8f;

#define NHEADS 16
#define EMBED  1024
#define HDIM   64
#define BSZ    2
#define SEQ    4096
#define MTOK   (BSZ * SEQ)   // 8192

#ifndef __has_builtin
#define __has_builtin(x) 0
#endif

#if __has_builtin(__builtin_amdgcn_global_load_async_to_lds_b128)
#define HAVE_ASYNC_LDS 1
#else
#define HAVE_ASYNC_LDS 0
#endif

// Copy 16 bytes global -> LDS. Async (ASYNCcnt) when available.
__device__ __forceinline__ void async_copy16(const void* g, void* l) {
#if HAVE_ASYNC_LDS
  typedef int v4i_ __attribute__((vector_size(16)));
  typedef __attribute__((address_space(1))) v4i_* gp_t;
  typedef __attribute__((address_space(3))) v4i_* lp_t;
  __builtin_amdgcn_global_load_async_to_lds_b128((gp_t)g, (lp_t)l, 0, 0);
#else
  *(uint4*)l = *(const uint4*)g;
#endif
}

__device__ __forceinline__ void async_join() {
#if HAVE_ASYNC_LDS
#if __has_builtin(__builtin_amdgcn_s_wait_asynccnt)
  __builtin_amdgcn_s_wait_asynccnt(0);
#else
  asm volatile("s_wait_asynccnt 0" ::: "memory");
#endif
#endif
}

__device__ __forceinline__ v8f wmma32(v16h a, v16h b, v8f c) {
  // D = A(16x32 f16) * B(32x16 f16) + C(16x16 f32)
  return __builtin_amdgcn_wmma_f32_16x16x32_f16(
      /*neg_a=*/false, a, /*neg_b=*/false, b,
      /*c_mod=*/(short)0, c, /*reuse_a=*/false, /*reuse_b=*/false);
}

// ---------------------------------------------------------------------------
// fp32 -> f16 cast (4-wide)
// ---------------------------------------------------------------------------
__global__ void cast_f32_to_f16_kernel(const float* __restrict__ in,
                                       _Float16* __restrict__ out, int n4) {
  int i = blockIdx.x * blockDim.x + threadIdx.x;
  if (i < n4) {
    float4 v = ((const float4*)in)[i];
    _Float16* o = out + (size_t)i * 4;
    o[0] = (_Float16)v.x; o[1] = (_Float16)v.y;
    o[2] = (_Float16)v.z; o[3] = (_Float16)v.w;
  }
}

// fp32 [R][C] -> f16 transposed [C][R] (LDS-tiled, both sides coalesced)
__global__ void __launch_bounds__(256)
castT_f32_to_f16_kernel(const float* __restrict__ in,
                        _Float16* __restrict__ out, int R, int C) {
  __shared__ float tile[32][33];
  const int c0 = blockIdx.x * 32, r0 = blockIdx.y * 32;
  const int tx = threadIdx.x & 31, ty = threadIdx.x >> 5;   // 32 x 8
  #pragma unroll
  for (int i = 0; i < 32; i += 8)
    tile[ty + i][tx] = in[(size_t)(r0 + ty + i) * C + c0 + tx];
  __syncthreads();
  #pragma unroll
  for (int i = 0; i < 32; i += 8)
    out[(size_t)(c0 + ty + i) * R + r0 + tx] = (_Float16)tile[tx][ty + i];
}

// int mask {0,1} -> f16 additive bias (-10000 * mask), 4-wide
__global__ void mask_to_f16_kernel(const int* __restrict__ in,
                                   _Float16* __restrict__ out, int n4) {
  int i = blockIdx.x * blockDim.x + threadIdx.x;
  if (i < n4) {
    int4 v = ((const int4*)in)[i];
    _Float16* o = out + (size_t)i * 4;
    o[0] = (_Float16)(-10000.0f * (float)v.x);
    o[1] = (_Float16)(-10000.0f * (float)v.y);
    o[2] = (_Float16)(-10000.0f * (float)v.z);
    o[3] = (_Float16)(-10000.0f * (float)v.w);
  }
}

// ---------------------------------------------------------------------------
// Tiled WMMA GEMM:  C[M,N] = A[M,K] @ B[K,N] + bias[N], B given TRANSPOSED
// (BmT = B^T, [N][K]).  128 threads (4 waves), block tile 128x64, wave tile
// 32x64 (each B fragment reused by 2 WMMAs), K step 32, async double-buffered.
// storeMode: 0 = f32 row-major, 1 = f16 [b,h,l,hd], 2 = f16 [b,h,hd,l].
// ---------------------------------------------------------------------------
__global__ void __launch_bounds__(128)
gemm_bias_wmma_kernel(const _Float16* __restrict__ A,
                      const _Float16* __restrict__ BmT,
                      const float* __restrict__ bias,
                      _Float16* __restrict__ outH,
                      float* __restrict__ outF,
                      int M, int N, int K, int storeMode) {
  // rows padded to 40 halves (80 B = 20 banks; 16 rows -> 16 distinct banks)
  __shared__ __align__(16) _Float16 As[2][128][40];  // [m][k]
  __shared__ __align__(16) _Float16 Bs[2][64][40];   // [n][k]

  const int t    = threadIdx.x;
  const int wave = t >> 5;
  const int lane = t & 31;
  const int half = lane >> 4;
  const int ln   = lane & 15;
  const int kbA  = half * 8;

  const int n0 = blockIdx.x * 64;
  const int m0 = blockIdx.y * 128;

  // tile fill: A 128 rows x 32 halves (1 row/thread, 4x16B);
  //            B 64 rows x 32 halves (t>>1 row, 2x16B)
  const int br = t >> 1, bc = (t & 1) * 16;
  auto load_tiles = [&](int kk0, int bufi) {
    const _Float16* sa = A + (size_t)(m0 + t) * K + kk0;
    async_copy16(sa,      &As[bufi][t][0]);
    async_copy16(sa + 8,  &As[bufi][t][8]);
    async_copy16(sa + 16, &As[bufi][t][16]);
    async_copy16(sa + 24, &As[bufi][t][24]);
    const _Float16* sb = BmT + (size_t)(n0 + br) * K + kk0 + bc;
    async_copy16(sb,     &Bs[bufi][br][bc]);
    async_copy16(sb + 8, &Bs[bufi][br][bc + 8]);
  };

  v8f acc[2][4] = {};
  load_tiles(0, 0);

  int it = 0;
  for (int k0 = 0; k0 < K; k0 += 32, ++it) {
    const int buf = it & 1;
    async_join();
    __syncthreads();
    if (k0 + 32 < K) load_tiles(k0 + 32, buf ^ 1);

    // batch all fragment loads, then issue 8 WMMAs back-to-back
    v16h a[2];
    #pragma unroll
    for (int s = 0; s < 2; ++s) {
      const _Float16* ap = &As[buf][wave * 32 + s * 16 + ln][0];
      #pragma unroll
      for (int i = 0; i < 8; ++i) {
        a[s][i]     = ap[2 * kbA + i];
        a[s][8 + i] = ap[2 * kbA + 8 + i];
      }
    }
    v16h b[4];
    #pragma unroll
    for (int j = 0; j < 4; ++j) {
      const _Float16* bp = &Bs[buf][j * 16 + ln][half * 16];
      #pragma unroll
      for (int i = 0; i < 16; ++i) b[j][i] = bp[i];
    }
    #pragma unroll
    for (int j = 0; j < 4; ++j) {
      acc[0][j] = wmma32(a[0], b[j], acc[0][j]);
      acc[1][j] = wmma32(a[1], b[j], acc[1][j]);
    }
  }

  // Epilogue: C layout = lane holds column ln, rows mb..mb+7 (consecutive)
  const int mb = half * 8;
  #pragma unroll
  for (int s = 0; s < 2; ++s) {
    #pragma unroll
    for (int j = 0; j < 4; ++j) {
      int col  = n0 + j * 16 + ln;
      float bv = bias[col];
      const int row0 = m0 + wave * 32 + s * 16 + mb;
      if (storeMode == 2) {
        // transposed head-split: [b,h,hd,l]; 8 consecutive l -> one 16B store
        int bb = row0 >> 12;
        int ll = row0 & (SEQ - 1);
        int hh = col >> 6;
        int dd = col & (HDIM - 1);
        _Float16 tmp[8];
        #pragma unroll
        for (int r = 0; r < 8; ++r) tmp[r] = (_Float16)(acc[s][j][r] + bv);
        _Float16* dst = outH + ((((size_t)bb * NHEADS + hh) * HDIM + dd) * SEQ + ll);
        *(uint4*)dst = *(const uint4*)tmp;
      } else {
        #pragma unroll
        for (int r = 0; r < 8; ++r) {
          int row   = row0 + r;
          float val = acc[s][j][r] + bv;
          if (storeMode == 1) {
            int bb = row >> 12;
            int ll = row & (SEQ - 1);
            int hh = col >> 6;
            int dd = col & (HDIM - 1);
            outH[((((size_t)bb * NHEADS + hh) * SEQ + ll) << 6) + dd] = (_Float16)val;
          } else {
            outF[(size_t)row * N + col] = val;
          }
        }
      }
    }
  }
}

// ---------------------------------------------------------------------------
// Flash attention: grid (L/64, B*H), block 128 (4 waves).
// K ([b,h,l,hd]) and V^T ([b,h,hd,l]) 32-key tiles staged in LDS via async
// copies (double-buffered); online softmax; merged-head f16 output.
// ---------------------------------------------------------------------------
__global__ void __launch_bounds__(128)
flash_attn_wmma_kernel(const _Float16* __restrict__ Q,
                       const _Float16* __restrict__ K,
                       const _Float16* __restrict__ Vt,
                       const _Float16* __restrict__ maskh,   // [L,L] = -1e4*mask
                       _Float16* __restrict__ O) {
  __shared__ __align__(16) _Float16 Ks[2][32][72];    // [key][d]
  __shared__ __align__(16) _Float16 Vts[2][64][40];   // [d][key]
  __shared__ __align__(16) _Float16 ps[4][16][32];    // per-wave P staging

  const int t    = threadIdx.x;
  const int wave = t >> 5;
  const int lane = t & 31;
  const int half = lane >> 4;
  const int ln   = lane & 15;
  const int kbA  = half * 8;
  const int mb   = half * 8;

  const int bh = blockIdx.y;                    // 0..31
  const int bb = bh >> 4;
  const int hh = bh & (NHEADS - 1);
  const int q0 = blockIdx.x * 64 + wave * 16;   // q position within L

  const size_t base = (size_t)bh * SEQ * HDIM;
  const _Float16* Kb  = K  + base;
  const _Float16* Vtb = Vt + base;              // [hd][l] within head

  // tile-fill: K 32x64 (row t>>2, col (t&3)*16); V^T 64x32 (row t>>1, col (t&1)*16)
  const int kr = t >> 2, kc2 = (t & 3) * 16;
  const int vr = t >> 1, vc  = (t & 1) * 16;
  auto load_kv = [&](int kcn, int bufi) {
    const _Float16* sk = Kb + (size_t)(kcn + kr) * HDIM + kc2;
    async_copy16(sk,     &Ks[bufi][kr][kc2]);
    async_copy16(sk + 8, &Ks[bufi][kr][kc2 + 8]);
    const _Float16* sv = Vtb + (size_t)vr * SEQ + kcn + vc;
    async_copy16(sv,     &Vts[bufi][vr][vc]);
    async_copy16(sv + 8, &Vts[bufi][vr][vc + 8]);
  };

  // Q fragments over d (two K=32 chunks), pre-scaled by 1/sqrt(hd) = 1/8
  v16h qf[2];
  {
    const _Float16* qp = Q + base + (size_t)(q0 + ln) * HDIM;
    const _Float16 sc = (_Float16)0.125f;
    #pragma unroll
    for (int c = 0; c < 2; ++c) {
      #pragma unroll
      for (int i = 0; i < 8; ++i) {
        qf[c][i]     = qp[c * 32 + 2 * kbA + i] * sc;
        qf[c][8 + i] = qp[c * 32 + 2 * kbA + 8 + i] * sc;
      }
    }
  }

  v8f acc[4] = {};
  float mrow[8], lrow[8];
  #pragma unroll
  for (int r = 0; r < 8; ++r) { mrow[r] = -1e30f; lrow[r] = 0.0f; }

  load_kv(0, 0);

  int it = 0;
  for (int kc = 0; kc < SEQ; kc += 32, ++it) {
    const int buf = it & 1;
    async_join();
    __syncthreads();
    if (kc + 32 < SEQ) load_kv(kc + 32, buf ^ 1);

    // ---- scores: S[16 q x 32 k] = (Q/8) @ K^T, two 16x16 col tiles ----
    v8f s01[2] = {};
    #pragma unroll
    for (int jt = 0; jt < 2; ++jt) {
      const _Float16* kp = &Ks[buf][jt * 16 + ln][0];
      v16h kf[2];   // batch both d-chunks, then 2 WMMAs back-to-back
      #pragma unroll
      for (int c = 0; c < 2; ++c) {
        const _Float16* kq = kp + c * 32 + half * 16;
        #pragma unroll
        for (int i = 0; i < 16; ++i) kf[c][i] = kq[i];
      }
      s01[jt] = wmma32(qf[0], kf[0], s01[jt]);
      s01[jt] = wmma32(qf[1], kf[1], s01[jt]);
    }

    // ---- mask bias + online softmax (rows live across 16-lane halves) ----
    float p0[8], p1[8], cmax[8];
    #pragma unroll
    for (int r = 0; r < 8; ++r) {
      const size_t mbase = (size_t)(q0 + mb + r) * SEQ + kc;
      float s0 = s01[0][r] + (float)maskh[mbase + ln];
      float s1 = s01[1][r] + (float)maskh[mbase + 16 + ln];
      p0[r] = s0; p1[r] = s1;
      cmax[r] = fmaxf(s0, s1);
    }
    #pragma unroll
    for (int off = 1; off < 16; off <<= 1) {
      #pragma unroll
      for (int r = 0; r < 8; ++r)
        cmax[r] = fmaxf(cmax[r], __shfl_xor(cmax[r], off));
    }
    float corr[8];
    #pragma unroll
    for (int r = 0; r < 8; ++r) {
      float mnew = fmaxf(mrow[r], cmax[r]);
      corr[r] = __expf(mrow[r] - mnew);
      mrow[r] = mnew;
      p0[r] = __expf(p0[r] - mnew);
      p1[r] = __expf(p1[r] - mnew);
    }
    float csum[8];
    #pragma unroll
    for (int r = 0; r < 8; ++r) csum[r] = p0[r] + p1[r];
    #pragma unroll
    for (int off = 1; off < 16; off <<= 1) {
      #pragma unroll
      for (int r = 0; r < 8; ++r) csum[r] += __shfl_xor(csum[r], off);
    }
    #pragma unroll
    for (int r = 0; r < 8; ++r) lrow[r] = lrow[r] * corr[r] + csum[r];
    #pragma unroll
    for (int j = 0; j < 4; ++j) {
      #pragma unroll
      for (int r = 0; r < 8; ++r) acc[j][r] *= corr[r];
    }

    // ---- P: C-layout -> LDS -> A-fragment (per-wave slice, LDS in-order) ----
    #pragma unroll
    for (int r = 0; r < 8; ++r) {
      ps[wave][mb + r][ln]      = (_Float16)p0[r];
      ps[wave][mb + r][16 + ln] = (_Float16)p1[r];
    }
    v16h pf;
    const _Float16* pp = &ps[wave][ln][0];
    #pragma unroll
    for (int i = 0; i < 8; ++i) {
      pf[i]     = pp[2 * kbA + i];
      pf[8 + i] = pp[2 * kbA + 8 + i];
    }

    // ---- O += P @ V: batch 4 V^T fragments, then 4 WMMAs back-to-back ----
    v16h vf[4];
    #pragma unroll
    for (int jn = 0; jn < 4; ++jn) {
      const _Float16* vp = &Vts[buf][jn * 16 + ln][half * 16];
      #pragma unroll
      for (int i = 0; i < 16; ++i) vf[jn][i] = vp[i];
    }
    #pragma unroll
    for (int jn = 0; jn < 4; ++jn) acc[jn] = wmma32(pf, vf[jn], acc[jn]);
  }

  // ---- normalize + merged-head store: O[(b*L + q)*E + h*64 + d] ----
  #pragma unroll
  for (int jn = 0; jn < 4; ++jn) {
    int d = jn * 16 + ln;
    #pragma unroll
    for (int r = 0; r < 8; ++r) {
      int ql   = q0 + mb + r;
      float ov = acc[jn][r] / lrow[r];
      O[(size_t)(bb * SEQ + ql) * EMBED + hh * HDIM + d] = (_Float16)ov;
    }
  }
}

// ---------------------------------------------------------------------------
// Host-side orchestration
// ---------------------------------------------------------------------------
extern "C" void kernel_launch(void* const* d_in, const int* in_sizes, int n_in,
                              void* d_out, int out_size, void* d_ws, size_t ws_size,
                              hipStream_t stream) {
  const float* x    = (const float*)d_in[0];
  const int*   mask = (const int*)d_in[1];
  const float* Wq   = (const float*)d_in[2];
  const float* bq   = (const float*)d_in[3];
  const float* Wk   = (const float*)d_in[4];
  const float* bk   = (const float*)d_in[5];
  const float* Wv   = (const float*)d_in[6];
  const float* bv   = (const float*)d_in[7];
  const float* Wo   = (const float*)d_in[8];
  const float* bo   = (const float*)d_in[9];
  float* out = (float*)d_out;

  // workspace carve-up (f16 staging buffers), 256B aligned
  char* ws = (char*)d_ws;
  size_t off = 0;
  auto carve = [&](size_t bytes) -> char* {
    char* p = ws + off;
    off += (bytes + 255) & ~(size_t)255;
    return p;
  };
  const size_t xN = (size_t)MTOK * EMBED;     // 8192*1024
  const size_t wN = (size_t)EMBED * EMBED;    // 1024*1024
  const size_t mN = (size_t)SEQ * SEQ;        // 4096*4096
  _Float16* xh    = (_Float16*)carve(xN * 2);
  _Float16* wqt   = (_Float16*)carve(wN * 2);   // Wq^T  [out][in]
  _Float16* wkt   = (_Float16*)carve(wN * 2);
  _Float16* wvt   = (_Float16*)carve(wN * 2);
  _Float16* wot   = (_Float16*)carve(wN * 2);
  _Float16* qh    = (_Float16*)carve(xN * 2);   // [b,h,l,hd]
  _Float16* kh    = (_Float16*)carve(xN * 2);   // [b,h,l,hd]
  _Float16* vth   = (_Float16*)carve(xN * 2);   // [b,h,hd,l]  (V transposed)
  _Float16* oh    = (_Float16*)carve(xN * 2);   // merged-head [b*l, E]
  _Float16* mh    = (_Float16*)carve(mN * 2);   // -1e4 * mask, f16
  (void)ws_size; (void)in_sizes; (void)n_in; (void)out_size;

  // 1) casts: x plain; weights cast + transpose; mask -> f16 bias
  cast_f32_to_f16_kernel<<<(int)(xN / 4 / 256), 256, 0, stream>>>(x, xh, (int)(xN / 4));
  dim3 tg(EMBED / 32, EMBED / 32);
  castT_f32_to_f16_kernel<<<tg, 256, 0, stream>>>(Wq, wqt, EMBED, EMBED);
  castT_f32_to_f16_kernel<<<tg, 256, 0, stream>>>(Wk, wkt, EMBED, EMBED);
  castT_f32_to_f16_kernel<<<tg, 256, 0, stream>>>(Wv, wvt, EMBED, EMBED);
  castT_f32_to_f16_kernel<<<tg, 256, 0, stream>>>(Wo, wot, EMBED, EMBED);
  mask_to_f16_kernel<<<(int)(mN / 4 / 256), 256, 0, stream>>>(mask, mh, (int)(mN / 4));

  // 2) Q/K/V projections (WMMA GEMM, fused bias + layout transform)
  dim3 gg(EMBED / 64, MTOK / 128);   // (16, 64)
  gemm_bias_wmma_kernel<<<gg, 128, 0, stream>>>(xh, wqt, bq, qh, nullptr,
                                                MTOK, EMBED, EMBED, 1);
  gemm_bias_wmma_kernel<<<gg, 128, 0, stream>>>(xh, wkt, bk, kh, nullptr,
                                                MTOK, EMBED, EMBED, 1);
  gemm_bias_wmma_kernel<<<gg, 128, 0, stream>>>(xh, wvt, bv, vth, nullptr,
                                                MTOK, EMBED, EMBED, 2);

  // 3) flash attention (never materializes the 2.1 GB score matrix)
  flash_attn_wmma_kernel<<<dim3(SEQ / 64, BSZ * NHEADS), 128, 0, stream>>>(
      qh, kh, vth, mh, oh);

  // 4) output projection, fp32 result
  gemm_bias_wmma_kernel<<<gg, 128, 0, stream>>>(oh, wot, bo, nullptr, out,
                                                MTOK, EMBED, EMBED, 0);
}
